// Attention_70720931496164
// MI455X (gfx1250) — matrix operand order repaired and checked by reading, not compiled, not run on hardware
//
#include <hip/hip_runtime.h>
#include <hip/hip_bf16.h>
#include <math.h>

// ---------------------------------------------------------------------------
// Bahdanau additive attention, fused flash-style for MI455X (gfx1250).
//   B=128, T=2048, D=256.  Dominant GEMM h@W2^T done with
//   v_wmma_f32_16x16x32_bf16 (f32 accumulate), one single pass over h (256MB)
//   with online softmax + running weighted-sum accumulation.
//   Round 3: DPP16 butterfly reduction (pure VALU) instead of ds_bpermute.
// ---------------------------------------------------------------------------

typedef __bf16 bf16_t;
typedef __attribute__((ext_vector_type(16))) bf16_t v16bf;
typedef __attribute__((ext_vector_type(8)))  bf16_t v8bf;
typedef __attribute__((ext_vector_type(8)))  float  v8f;

#define BB     128
#define TT     2048
#define DD     256
#define NSPLIT 4
#define TSPL   (TT / NSPLIT)    // 512 rows of T per workgroup
#define NTILES (TSPL / 16)      // 32 tiles of 16 rows
#define LDSP   264              // padded bf16 row stride (528B = 33*16B, kills bank conflicts)
#define NEGV   (-1000000000.0f)

__device__ __forceinline__ float fast_tanhf(float x) {
#if __has_builtin(__builtin_amdgcn_tanhf)
  return __builtin_amdgcn_tanhf(x);        // native v_tanh_f32 on gfx1250
#else
  float e = __expf(fminf(fmaxf(2.0f * x, -40.0f), 40.0f));
  return (e - 1.0f) * __builtin_amdgcn_rcpf(e + 1.0f);
#endif
}

// Sum across each 16-lane row, pure VALU (DPP16), no LDS round trips.
// xor1, xor2 complete each quad; half-row mirror pairs quads; row mirror
// pairs 8-groups (valid since quads are uniform after the first two steps).
__device__ __forceinline__ float red16_add(float w) {
  w += __int_as_float(__builtin_amdgcn_mov_dpp(__float_as_int(w), 0xB1, 0xf, 0xf, true));   // quad_perm(1,0,3,2)
  w += __int_as_float(__builtin_amdgcn_mov_dpp(__float_as_int(w), 0x4E, 0xf, 0xf, true));   // quad_perm(2,3,0,1)
  w += __int_as_float(__builtin_amdgcn_mov_dpp(__float_as_int(w), 0x141, 0xf, 0xf, true));  // row_half_mirror
  w += __int_as_float(__builtin_amdgcn_mov_dpp(__float_as_int(w), 0x140, 0xf, 0xf, true));  // row_mirror
  return w;
}

// ---------------------------------------------------------------------------
// Prep: W2 (f32, row-major [e][d]) -> bf16 copy (128KB, stays hot in L2)
// ---------------------------------------------------------------------------
__global__ __launch_bounds__(256) void k_w2_to_bf16(const float* __restrict__ W2,
                                                    bf16_t* __restrict__ W2bf) {
  int i = blockIdx.x * 256 + threadIdx.x;
  W2bf[i] = (bf16_t)W2[i];
}

// ---------------------------------------------------------------------------
// Prep: comb[b,e] = sum_k d[b,k]*W1[e,k] + b1[e] + b2[e]   (16.8 MFLOP, f32)
// ---------------------------------------------------------------------------
__global__ __launch_bounds__(256) void k_comb(const float* __restrict__ d,
                                              const float* __restrict__ W1,
                                              const float* __restrict__ b1,
                                              const float* __restrict__ b2,
                                              float* __restrict__ comb) {
  __shared__ float sd[DD];
  const int b = blockIdx.x;
  const int e = threadIdx.x;
  sd[e] = d[b * DD + e];
  __syncthreads();
  const float* wr = W1 + (size_t)e * DD;
  float acc = b1[e] + b2[e];
#pragma unroll 4
  for (int k = 0; k < DD; k += 4) {
    float4 w4 = *(const float4*)(wr + k);
    acc += sd[k] * w4.x + sd[k + 1] * w4.y + sd[k + 2] * w4.z + sd[k + 3] * w4.w;
  }
  comb[b * DD + e] = acc;
}

// ---------------------------------------------------------------------------
// Main fused kernel: one workgroup per (b, T-split).
//  - double-buffered bf16 h tiles in LDS; next tile's global loads issued
//    before the WMMA block so HBM latency overlaps matrix compute
//  - W2 B-fragments hoisted to VGPRs (16 frags x 8 VGPR = 128 VGPR/lane)
//  - 16 v_wmma_f32_16x16x32_bf16 per tile (16 x 256 x K=256 GEMM slice)
//  - tanh -> dot v -> DPP reduction -> mask -> online softmax
//  - running acc[e] += p[t]*h[t,e] from the same LDS tile (h read ONCE)
// ---------------------------------------------------------------------------
__global__ __launch_bounds__(256) void k_attn_main(
    const float* __restrict__ h, const unsigned char* __restrict__ mask,
    const float* __restrict__ comb, const bf16_t* __restrict__ W2bf,
    const float* __restrict__ v, float* __restrict__ u_out /*[B,T]*/,
    float* __restrict__ macc /*[B*NSPLIT,DD]*/, float* __restrict__ pml /*[B*NSPLIT,2]*/) {
  __shared__ bf16_t hA[2][16 * LDSP];  // double-buffered bf16 h tiles
  __shared__ float part[16][9];        // per-wave partial u contributions (padded)
  __shared__ float u_tile[16];

  const int tid  = threadIdx.x;
  const int wave = tid >> 5;   // 8 waves
  const int L    = tid & 31;
  const int half = L >> 4;     // 0: lanes 0-15, 1: lanes 16-31
  const int l15  = L & 15;

  const int b  = blockIdx.x / NSPLIT;
  const int s  = blockIdx.x % NSPLIT;
  const int t0base = s * TSPL;

  // --- hoist B-matrix (W2^T) fragments + per-lane bias/v into registers ---
  // Wave owns e-tiles n0 = (2*wave+et)*16.  B 32x16 bf16 as loaded (contiguous k, same as A): lane = N,
  // lanes 0-15 hold K=kb..kb+15, lanes 16-31 hold K=kb+16..kb+31 (contig in W2 row).
  v16bf bfrag[2][8];
  float combE[2], vE[2];
#pragma unroll
  for (int et = 0; et < 2; ++et) {
    const int e = (wave * 2 + et) * 16 + l15;
    combE[et] = comb[b * DD + e];
    vE[et]    = v[e];
    const bf16_t* wrow = W2bf + (size_t)e * DD;
#pragma unroll
    for (int c = 0; c < 8; ++c) {
      bfrag[et][c] = *(const v16bf*)(wrow + c * 32 + half * 16);
    }
  }

  float m_run = -INFINITY, l_run = 0.0f, acc = 0.0f;
  const float* hb = h + (size_t)b * TT * DD;

  // --- prologue: stage tile 0 ---
  {
    const float* src = hb + (size_t)t0base * DD;
#pragma unroll
    for (int j = 0; j < 16; ++j) hA[0][j * LDSP + tid] = (bf16_t)src[j * DD + tid];
  }
  __syncthreads();

  for (int tt = 0; tt < NTILES; ++tt) {
    const int cur = tt & 1, nxt = cur ^ 1;
    const int t0 = t0base + tt * 16;
    const bool have_next = (tt + 1 < NTILES);

    // issue next tile's global loads now; they overlap the 16 WMMAs below
    float stage[16];
    if (have_next) {
      const float* src = hb + (size_t)(t0 + 16) * DD;
#pragma unroll
      for (int j = 0; j < 16; ++j) stage[j] = src[j * DD + tid];
      if (tt + 2 < NTILES)  // prefetch two tiles ahead (global_prefetch_b8)
        __builtin_prefetch(src + 16 * DD + tid * 16, 0, 3);
    }

    // --- GEMM: each wave computes two 16x16 tiles of tanh-arg ---
    v8f c0 = {0.f, 0.f, 0.f, 0.f, 0.f, 0.f, 0.f, 0.f};
    v8f c1 = {0.f, 0.f, 0.f, 0.f, 0.f, 0.f, 0.f, 0.f};
    const bf16_t* arow = &hA[cur][l15 * LDSP];  // A 16x32 bf16: lane m=l15, K runs of 8
#pragma unroll
    for (int c = 0; c < 8; ++c) {
      const int kbA = c * 32 + half * 16;
      union { v16bf v; v8bf h8[2]; } a;
      a.h8[0] = *(const v8bf*)(arow + kbA);        // K = kbA .. kbA+7
      a.h8[1] = *(const v8bf*)(arow + kbA + 8);    // K = kbA+8 .. kbA+15
      c0 = __builtin_amdgcn_wmma_f32_16x16x32_bf16(false, a.v, false, bfrag[0][c],
                                                   (short)0, c0, false, false);
      c1 = __builtin_amdgcn_wmma_f32_16x16x32_bf16(false, a.v, false, bfrag[1][c],
                                                   (short)0, c1, false, false);
    }

    // --- tanh, *v[e], reduce over e (DPP butterfly within 16-lane rows) ---
#pragma unroll
    for (int r = 0; r < 8; ++r) {  // C layout: VGPR r -> t = r + 8*half, e = n0+l15
      float w = fast_tanhf(c0[r] + combE[0]) * vE[0] +
                fast_tanhf(c1[r] + combE[1]) * vE[1];
      w = red16_add(w);
      if (l15 == 0) part[r + half * 8][wave] = w;
    }
    __syncthreads();

    if (tid < 16) {  // fixed-order sum over 8 waves -> u; apply mask; emit raw u
      float su = 0.0f;
#pragma unroll
      for (int c2 = 0; c2 < 8; ++c2) su += part[tid][c2];
      if (mask[(size_t)b * TT + t0 + tid]) su = NEGV;
      u_tile[tid] = su;
      u_out[(size_t)b * TT + t0 + tid] = su;  // finalized to softmax later
    }
    __syncthreads();

    // --- online softmax update, redundant per thread (thread owns acc[e=tid]) ---
    float ut[16], mt = -INFINITY;
#pragma unroll
    for (int j = 0; j < 16; ++j) { ut[j] = u_tile[j]; mt = fmaxf(mt, ut[j]); }
    const float m_new = fmaxf(m_run, mt);
    const float scale = __expf(m_run - m_new);
    float psum = 0.0f;
    acc *= scale;
#pragma unroll
    for (int j = 0; j < 16; ++j) {
      float p = __expf(ut[j] - m_new);
      psum += p;
      acc += p * (float)hA[cur][j * LDSP + tid];  // h reused straight from LDS
    }
    l_run = l_run * scale + psum;
    m_run = m_new;

    // --- commit staged registers into the other buffer for the next tile ---
    if (have_next) {
#pragma unroll
      for (int j = 0; j < 16; ++j) hA[nxt][j * LDSP + tid] = (bf16_t)stage[j];
    }
    __syncthreads();
  }

  macc[(size_t)blockIdx.x * DD + tid] = acc;  // unnormalized partial
  if (tid == 0) { pml[blockIdx.x * 2 + 0] = m_run; pml[blockIdx.x * 2 + 1] = l_run; }
}

// ---------------------------------------------------------------------------
// Merge the NSPLIT T-splits per batch: h_prime and final (m,l) for the a-pass.
// ---------------------------------------------------------------------------
__global__ __launch_bounds__(256) void k_combine(const float* __restrict__ macc,
                                                 const float* __restrict__ pml,
                                                 float* __restrict__ hp,
                                                 float* __restrict__ ml) {
  const int b = blockIdx.x, e = threadIdx.x;
  float m[NSPLIT], l[NSPLIT], M = -INFINITY;
#pragma unroll
  for (int s = 0; s < NSPLIT; ++s) {
    m[s] = pml[(b * NSPLIT + s) * 2 + 0];
    l[s] = pml[(b * NSPLIT + s) * 2 + 1];
    M = fmaxf(M, m[s]);
  }
  float lt = 0.0f, a = 0.0f;
#pragma unroll
  for (int s = 0; s < NSPLIT; ++s) {
    const float sc = __expf(m[s] - M);
    lt += l[s] * sc;
    a  += sc * macc[(size_t)(b * NSPLIT + s) * DD + e];
  }
  hp[b * DD + e] = a / lt;
  if (e == 0) { ml[b * 2 + 0] = M; ml[b * 2 + 1] = lt; }
}

// ---------------------------------------------------------------------------
// In-place: a[b,t] = exp(u[b,t] - M[b]) / l[b]   (masked u=NEG -> exp -> 0)
// ---------------------------------------------------------------------------
__global__ __launch_bounds__(256) void k_finalize(float* __restrict__ a,
                                                  const float* __restrict__ ml) {
  const int i = blockIdx.x * 256 + threadIdx.x;
  const int b = i / TT;
  a[i] = __expf(a[i] - ml[b * 2 + 0]) / ml[b * 2 + 1];
}

// ---------------------------------------------------------------------------
extern "C" void kernel_launch(void* const* d_in, const int* in_sizes, int n_in,
                              void* d_out, int out_size, void* d_ws, size_t ws_size,
                              hipStream_t stream) {
  (void)in_sizes; (void)n_in; (void)out_size; (void)ws_size;
  const float*         d    = (const float*)d_in[0];
  const float*         h    = (const float*)d_in[1];
  const unsigned char* mask = (const unsigned char*)d_in[2];  // jnp.bool_: 1 byte/elt
  const float*         W1   = (const float*)d_in[3];
  const float*         b1   = (const float*)d_in[4];
  const float*         W2   = (const float*)d_in[5];
  const float*         b2   = (const float*)d_in[6];
  const float*         v    = (const float*)d_in[7];

  float* hp   = (float*)d_out;        // h_prime [B,DD]
  float* aout = hp + BB * DD;         // a       [B,T] (holds raw u until finalize)

  // workspace layout (~0.76 MB)
  char*   ws    = (char*)d_ws;
  float*  comb  = (float*)(ws);                 // B*DD         = 128 KB
  bf16_t* W2bf  = (bf16_t*)(ws + 131072);       // DD*DD bf16   = 128 KB
  float*  macc  = (float*)(ws + 262144);        // B*NSPLIT*DD  = 512 KB
  float*  pml   = (float*)(ws + 786432);        // B*NSPLIT*2   =   4 KB
  float*  ml    = (float*)(ws + 790528);        // B*2          =   1 KB

  k_w2_to_bf16<<<(DD * DD) / 256, 256, 0, stream>>>(W2, W2bf);
  k_comb<<<BB, 256, 0, stream>>>(d, W1, b1, b2, comb);
  k_attn_main<<<BB * NSPLIT, 256, 0, stream>>>(h, mask, comb, W2bf, v, aout, macc, pml);
  k_combine<<<BB, 256, 0, stream>>>(macc, pml, hp, ml);
  k_finalize<<<(BB * TT) / 256, 256, 0, stream>>>(aout, ml);
}
